// Model_65773129171099
// MI455X (gfx1250) — compile-verified, matrix operand order checked by reading, not run on hardware
//
#include <hip/hip_runtime.h>
#include <math.h>

// ---------------- problem constants (match reference) ----------------
#define NNODE 131072
#define HDIM  256
#define SG    64
#define GG    (NNODE / SG)        // 2048 graphs
#define NHEAD 8
#define DHEAD 32
#define FFD   512
#define FNODE 32
#define WALKL 20
#define DSEC  64
#define NEDGE (4 * NNODE)         // 524288
#define EPG   (NEDGE / GG)        // 256 edges per graph (edge list is graph-contiguous)
#define LAYERS 6
#define LNEPS 1e-5f

// ---------------- WMMA types / helpers (CDNA5 gfx1250, wave32) ----------------
typedef __attribute__((ext_vector_type(16))) __bf16        bf16x16;
typedef __attribute__((ext_vector_type(16))) unsigned short u16x16;
typedef __attribute__((ext_vector_type(8)))  float          f32x8;

__device__ __forceinline__ unsigned short f2bf_u(float x) {
  unsigned u = __builtin_bit_cast(unsigned, x);
  unsigned r = u + 0x7FFFu + ((u >> 16) & 1u);   // round-to-nearest-even
  return (unsigned short)(r >> 16);
}

__device__ __forceinline__ f32x8 wmma_bf16(bf16x16 a, bf16x16 b, f32x8 c) {
  // D = A(16x32 bf16) * B(32x16 bf16) + C(16x16 f32)
  return __builtin_amdgcn_wmma_f32_16x16x32_bf16(false, a, false, b, (short)0, c, false, false);
}

__device__ __forceinline__ f32x8 zero8() {
  f32x8 z = {0.f, 0.f, 0.f, 0.f, 0.f, 0.f, 0.f, 0.f};
  return z;
}

// A-fragment from LDS bf16 row-major [rows][ld]: lane&15 = row in 16-row block,
// (lane>>4)*8 selects K sub-block; halves 0..7 -> K+0..7, halves 8..15 -> K+16..23.
__device__ __forceinline__ bf16x16 fragA_lds(const unsigned short* base, int ld,
                                             int row0, int k0, int lane) {
  const unsigned short* p = base + (row0 + (lane & 15)) * ld + k0 + ((lane >> 4) << 3);
  u16x16 u;
#pragma unroll
  for (int i = 0; i < 8; ++i) { u[i] = p[i]; u[i + 8] = p[i + 16]; }
  return __builtin_bit_cast(bf16x16, u);
}

// B-fragment from pre-transposed bf16 weights Wt[Ncols][K] (row-major): lane&15 = column.
// Contiguous 16B + 16B per lane -> vectorizes to global_load_b128 pairs.
__device__ __forceinline__ bf16x16 fragB_bf(const unsigned short* __restrict__ Wt, int ldk,
                                            int k0, int col0, int lane) {
  const unsigned short* p = Wt + (size_t)(col0 + (lane & 15)) * ldk + k0 + ((lane >> 4) << 3);
  u16x16 u;
#pragma unroll
  for (int i = 0; i < 8; ++i) { u[i] = p[i]; u[i + 8] = p[i + 16]; }
  return __builtin_bit_cast(bf16x16, u);
}

// B-fragment gathered from LDS bf16 [K][ld] (row-major), lane&15 = column.
__device__ __forceinline__ bf16x16 fragB_lds(const unsigned short* base, int ld,
                                             int k0, int col0, int lane) {
  int c = col0 + (lane & 15);
  int kb = k0 + ((lane >> 4) << 3);
  u16x16 u;
#pragma unroll
  for (int i = 0; i < 8; ++i) { u[i] = base[(kb + i) * ld + c]; u[i + 8] = base[(kb + 16 + i) * ld + c]; }
  return __builtin_bit_cast(bf16x16, u);
}

// Stage a 64x256 f32 tile from global into LDS bf16 (row pitch ldd halves), float4 path.
__device__ __forceinline__ void stage_tile_bf16(const float* __restrict__ src,
                                                unsigned short* dst, int ldd, int tid) {
  const float4* s4 = (const float4*)src;
  for (int i = tid; i < (SG * HDIM) / 4; i += 256) {
    int r = i >> 6;                 // 64 float4 per row
    int c4 = (i & 63) << 2;
    float4 v = s4[i];
    unsigned lo = (unsigned)f2bf_u(v.x) | ((unsigned)f2bf_u(v.y) << 16);
    unsigned hi = (unsigned)f2bf_u(v.z) | ((unsigned)f2bf_u(v.w) << 16);
    unsigned* d = (unsigned*)(dst + r * ldd + c4);
    d[0] = lo; d[1] = hi;
  }
}

// ---------------- weight pre-convert + transpose: W[L][K][N] f32 -> Wt[L][N][K] bf16 ----------------
__global__ void wtconv_kernel(const float* __restrict__ W, unsigned short* __restrict__ Wt,
                              int K, int Ncols) {
  size_t idx = (size_t)blockIdx.x * 256 + threadIdx.x;   // over L*K*N
  size_t per = (size_t)K * Ncols;
  size_t l = idx / per;
  size_t rem = idx - l * per;
  int k = (int)(rem / Ncols);
  int n = (int)(rem - (size_t)k * Ncols);
  Wt[l * per + (size_t)n * K + k] = f2bf_u(W[idx]);
}

// ---------------- degree / dinv ----------------
__global__ void deg_init_kernel(float* d) {
  size_t i = (size_t)blockIdx.x * 256 + threadIdx.x;
  d[i] = 1.0f;                                    // self-loop
}
__global__ void deg_acc_kernel(const int* __restrict__ dst, float* d) {
  size_t e = (size_t)blockIdx.x * 256 + threadIdx.x;
  atomicAdd(&d[dst[e]], 1.0f);
}
__global__ void deg_fin_kernel(float* d) {
  size_t i = (size_t)blockIdx.x * 256 + threadIdx.x;
  d[i] = rsqrtf(d[i]);
}

// ---------------- encoder: h0 = [x@Wx+bx | (rwse*g+b)@Wse+bse] ----------------
__global__ void encoder_kernel(const float* __restrict__ x, const float* __restrict__ rwse,
                               const float* __restrict__ exw, const float* __restrict__ exb,
                               const float* __restrict__ bng, const float* __restrict__ bnb,
                               const float* __restrict__ esw, const float* __restrict__ esb,
                               float* __restrict__ A) {
  size_t idx = (size_t)blockIdx.x * 256 + threadIdx.x;   // over N*H
  int c = (int)(idx & 255);
  size_t n = idx >> 8;
  float v;
  if (c < (HDIM - DSEC)) {
    float s = exb[c];
#pragma unroll
    for (int w = 0; w < FNODE; ++w) s += x[n * FNODE + w] * exw[w * (HDIM - DSEC) + c];
    v = s;
  } else {
    int c2 = c - (HDIM - DSEC);
    float s = esb[c2];
#pragma unroll
    for (int w = 0; w < WALKL; ++w)
      s += (rwse[n * WALKL + w] * bng[w] + bnb[w]) * esw[w * DSEC + c2];
    v = s;
  }
  A[idx] = v;
}

// ---------------- 64x256 tile GEMM: out = in[Nx256] @ W (Wt bf16 [256cols][256k]) ----------------
__global__ __launch_bounds__(256) void gemm_h_kernel(const float* __restrict__ in,
                                                     const unsigned short* __restrict__ Wt,
                                                     float* __restrict__ out) {
  __shared__ unsigned short At[64 * 264];
  int tid = threadIdx.x;
  size_t rowbase = (size_t)blockIdx.x * 64;
  __builtin_prefetch(Wt + (size_t)tid * 256, 0, 1);   // global_prefetch_b8: warm weight panel
  stage_tile_bf16(in + rowbase * HDIM, At, 264, tid);
  __syncthreads();
  int wave = tid >> 5, lane = tid & 31;
  int col0 = wave * 32;
  f32x8 acc[4][2];
#pragma unroll
  for (int rt = 0; rt < 4; ++rt) { acc[rt][0] = zero8(); acc[rt][1] = zero8(); }
  for (int kc = 0; kc < 8; ++kc) {
    int kb = kc * 32;
    bf16x16 b0 = fragB_bf(Wt, HDIM, kb, col0, lane);
    bf16x16 b1 = fragB_bf(Wt, HDIM, kb, col0 + 16, lane);
#pragma unroll
    for (int rt = 0; rt < 4; ++rt) {
      bf16x16 a = fragA_lds(At, 264, rt * 16, kb, lane);
      acc[rt][0] = wmma_bf16(a, b0, acc[rt][0]);
      acc[rt][1] = wmma_bf16(a, b1, acc[rt][1]);
    }
  }
#pragma unroll
  for (int rt = 0; rt < 4; ++rt)
#pragma unroll
    for (int j = 0; j < 2; ++j) {
      int col = col0 + j * 16 + (lane & 15);
#pragma unroll
      for (int r = 0; r < 8; ++r) {
        int row = rt * 16 + ((lane >> 4) << 3) + r;
        out[(rowbase + row) * HDIM + col] = acc[rt][j][r];
      }
    }
}

// ---------------- GCN aggregation + bias + residual + graph LayerNorm ----------------
__global__ __launch_bounds__(256) void gcn_agg_kernel(const float* __restrict__ hw,
                                                      const float* __restrict__ hres,
                                                      const int* __restrict__ esrc,
                                                      const int* __restrict__ edst,
                                                      const float* __restrict__ dinv,
                                                      const float* __restrict__ gcnb,
                                                      const float* __restrict__ cng,
                                                      const float* __restrict__ cnb,
                                                      float* __restrict__ outB) {
  __shared__ float hwT[64 * 256];
  __shared__ float accT[64 * 256];
  __shared__ float coef[EPG];
  __shared__ unsigned char sl[EPG], dl[EPG];
  __shared__ float selfc[64];
  __shared__ float redA[256], redB[256];
  __shared__ float muv[2];

  int tid = threadIdx.x;
  int g = blockIdx.x;
  size_t nb = (size_t)g * SG;

  {
    const float4* s4 = (const float4*)(hw + nb * HDIM);
    float4 z = {0.f, 0.f, 0.f, 0.f};
    for (int i = tid; i < (SG * HDIM) / 4; i += 256) {
      ((float4*)hwT)[i] = s4[i];
      ((float4*)accT)[i] = z;
    }
  }
  {
    int e = g * EPG + tid;
    int s = esrc[e], d = edst[e];
    sl[tid] = (unsigned char)(s & 63);
    dl[tid] = (unsigned char)(d & 63);
    coef[tid] = dinv[s] * dinv[d];
  }
  if (tid < SG) { float di = dinv[nb + tid]; selfc[tid] = di * di; }
  __syncthreads();

  // channel-per-thread scatter (conflict-free across threads)
  int c = tid;
  for (int e = 0; e < EPG; ++e) accT[dl[e] * 256 + c] += hwT[sl[e] * 256 + c] * coef[e];
  for (int n = 0; n < SG; ++n)  accT[n * 256 + c]     += hwT[n * 256 + c] * selfc[n];
  __syncthreads();

  float s1 = 0.f, s2 = 0.f;
  for (int i = tid; i < SG * HDIM; i += 256) {
    int ch = i & 255;
    float v = accT[i] + gcnb[ch] + hres[nb * HDIM + i];
    accT[i] = v;
    s1 += v; s2 += v * v;
  }
  redA[tid] = s1; redB[tid] = s2;
  __syncthreads();
  for (int off = 128; off > 0; off >>= 1) {
    if (tid < off) { redA[tid] += redA[tid + off]; redB[tid] += redB[tid + off]; }
    __syncthreads();
  }
  if (tid == 0) {
    float mu = redA[0] / (float)(SG * HDIM);
    float var = redB[0] / (float)(SG * HDIM) - mu * mu;
    muv[0] = mu; muv[1] = rsqrtf(var + LNEPS);
  }
  __syncthreads();
  float mu = muv[0], rstd = muv[1];
  for (int i = tid; i < SG * HDIM; i += 256) {
    int ch = i & 255;
    outB[nb * HDIM + i] = (accT[i] - mu) * rstd * cng[ch] + cnb[ch];
  }
}

// ---------------- per-WG 256x256 projection into LDS bf16 tile ----------------
__device__ __forceinline__ void wg_proj_256(const unsigned short* inT,
                                            const unsigned short* __restrict__ Wt,
                                            const float* __restrict__ bias, unsigned short* outT,
                                            int wave, int lane) {
  int col0 = wave * 32;
  f32x8 acc[4][2];
#pragma unroll
  for (int rt = 0; rt < 4; ++rt) { acc[rt][0] = zero8(); acc[rt][1] = zero8(); }
  for (int kc = 0; kc < 8; ++kc) {
    int kb = kc * 32;
    bf16x16 b0 = fragB_bf(Wt, HDIM, kb, col0, lane);
    bf16x16 b1 = fragB_bf(Wt, HDIM, kb, col0 + 16, lane);
#pragma unroll
    for (int rt = 0; rt < 4; ++rt) {
      bf16x16 a = fragA_lds(inT, 264, rt * 16, kb, lane);
      acc[rt][0] = wmma_bf16(a, b0, acc[rt][0]);
      acc[rt][1] = wmma_bf16(a, b1, acc[rt][1]);
    }
  }
#pragma unroll
  for (int rt = 0; rt < 4; ++rt)
#pragma unroll
    for (int j = 0; j < 2; ++j) {
      int col = col0 + j * 16 + (lane & 15);
      float bv = bias[col];
#pragma unroll
      for (int r = 0; r < 8; ++r) {
        int row = rt * 16 + ((lane >> 4) << 3) + r;
        outT[row * 264 + col] = f2bf_u(acc[rt][j][r] + bv);
      }
    }
}

// ---------------- fused per-graph MHA (+residual, + full-LN statistics) ----------------
__global__ __launch_bounds__(256) void mha_kernel(const float* __restrict__ h,
                                                  const unsigned short* __restrict__ qwt,
                                                  const float* __restrict__ qb,
                                                  const unsigned short* __restrict__ kwt,
                                                  const float* __restrict__ kbv,
                                                  const unsigned short* __restrict__ owt,
                                                  const float* __restrict__ ob,
                                                  float* __restrict__ outC, double* __restrict__ scal) {
  __shared__ __align__(16) unsigned char smem[129280];
  unsigned short* Ht = (unsigned short*)smem;               // [64][264] bf16 (h tile, reused as AV)
  unsigned short* Qt = (unsigned short*)(smem + 33792);     // [64][264]
  unsigned short* Kt = (unsigned short*)(smem + 67584);     // [64][264]  (V = K quirk)
  float*          Sc = (float*)(smem + 101376);             // [64][65]
  unsigned short* Pt = (unsigned short*)(smem + 118016);    // [64][68]
  float*        redA = (float*)(smem + 126720);             // [256]
  float*        redB = redA + 256;                          // [256]

  int tid = threadIdx.x;
  int wave = tid >> 5, lane = tid & 31;
  size_t rowbase = (size_t)blockIdx.x * SG;

  __builtin_prefetch(qwt + (size_t)tid * 256, 0, 1);
  __builtin_prefetch(kwt + (size_t)tid * 256, 0, 1);
  stage_tile_bf16(h + rowbase * HDIM, Ht, 264, tid);
  __syncthreads();
  wg_proj_256(Ht, qwt, qb, Qt, wave, lane);
  __syncthreads();
  wg_proj_256(Ht, kwt, kbv, Kt, wave, lane);
  __syncthreads();

  const float scale = 0.17677669529663687f;  // 1/sqrt(DHEAD)
  for (int hh = 0; hh < NHEAD; ++hh) {
    // scores S = Q_h K_h^T : 16 tiles, 2 per wave (K^T fragment == A-style load of Kt)
    for (int t = wave; t < 16; t += 8) {
      int rt = t >> 2, ct = t & 3;
      f32x8 s = zero8();
      bf16x16 aq = fragA_lds(Qt, 264, rt * 16, hh * DHEAD, lane);
      bf16x16 bk = fragA_lds(Kt, 264, ct * 16, hh * DHEAD, lane);
      s = wmma_bf16(aq, bk, s);
#pragma unroll
      for (int r = 0; r < 8; ++r) {
        int row = rt * 16 + ((lane >> 4) << 3) + r;
        Sc[row * 65 + ct * 16 + (lane & 15)] = s[r];
      }
    }
    __syncthreads();
    if (tid < SG) {
      float m = -1e30f;
      for (int j = 0; j < SG; ++j) { float v = Sc[tid * 65 + j] * scale; m = fmaxf(m, v); }
      float ssum = 0.f;
      for (int j = 0; j < SG; ++j) {
        float e = __expf(Sc[tid * 65 + j] * scale - m);
        Sc[tid * 65 + j] = e; ssum += e;
      }
      float inv = 1.f / ssum;
      for (int j = 0; j < SG; ++j) Pt[tid * 68 + j] = f2bf_u(Sc[tid * 65 + j] * inv);
    }
    __syncthreads();
    // AV = P @ V  (V = K), 8 tiles, one per wave; write into Ht (h tile no longer needed)
    {
      int rt = wave >> 1, ct = wave & 1;
      f32x8 acc = zero8();
#pragma unroll
      for (int kc = 0; kc < 2; ++kc) {
        bf16x16 a = fragA_lds(Pt, 68, rt * 16, kc * 32, lane);
        bf16x16 b = fragB_lds(Kt, 264, kc * 32, hh * DHEAD + ct * 16, lane);
        acc = wmma_bf16(a, b, acc);
      }
#pragma unroll
      for (int r = 0; r < 8; ++r) {
        int row = rt * 16 + ((lane >> 4) << 3) + r;
        Ht[row * 264 + hh * DHEAD + ct * 16 + (lane & 15)] = f2bf_u(acc[r]);
      }
    }
    __syncthreads();
  }

  // O projection + residual; accumulate sum/sumsq for full-LN
  float s1 = 0.f, s2 = 0.f;
  {
    int col0 = wave * 32;
    f32x8 acc[4][2];
#pragma unroll
    for (int rt = 0; rt < 4; ++rt) { acc[rt][0] = zero8(); acc[rt][1] = zero8(); }
    for (int kc = 0; kc < 8; ++kc) {
      int kb = kc * 32;
      bf16x16 b0 = fragB_bf(owt, HDIM, kb, col0, lane);
      bf16x16 b1 = fragB_bf(owt, HDIM, kb, col0 + 16, lane);
#pragma unroll
      for (int rt = 0; rt < 4; ++rt) {
        bf16x16 a = fragA_lds(Ht, 264, rt * 16, kb, lane);
        acc[rt][0] = wmma_bf16(a, b0, acc[rt][0]);
        acc[rt][1] = wmma_bf16(a, b1, acc[rt][1]);
      }
    }
#pragma unroll
    for (int rt = 0; rt < 4; ++rt)
#pragma unroll
      for (int j = 0; j < 2; ++j) {
        int col = col0 + j * 16 + (lane & 15);
        float bv = ob[col];
#pragma unroll
        for (int r = 0; r < 8; ++r) {
          int row = rt * 16 + ((lane >> 4) << 3) + r;
          float v = acc[rt][j][r] + bv + h[(rowbase + row) * HDIM + col];
          outC[(rowbase + row) * HDIM + col] = v;
          s1 += v; s2 += v * v;
        }
      }
  }
  redA[tid] = s1; redB[tid] = s2;
  __syncthreads();
  for (int off = 128; off > 0; off >>= 1) {
    if (tid < off) { redA[tid] += redA[tid + off]; redB[tid] += redB[tid + off]; }
    __syncthreads();
  }
  if (tid == 0) { atomicAdd(&scal[0], (double)redA[0]); atomicAdd(&scal[1], (double)redB[0]); }
}

// ---------------- full-LN combiners ----------------
__global__ void zero2_kernel(double* s) {
  if (threadIdx.x == 0 && blockIdx.x == 0) { s[0] = 0.0; s[1] = 0.0; }
}

__global__ void combine_attn_kernel(const float* __restrict__ Bg, const float* __restrict__ Cg,
                                    const double* __restrict__ scal, const float* __restrict__ gma,
                                    const float* __restrict__ bta, float* __restrict__ Ag) {
  size_t idx = (size_t)blockIdx.x * 256 + threadIdx.x;
  int ch = (int)(idx & 255);
  double M = (double)NNODE * (double)HDIM;
  double mu = scal[0] / M;
  double var = scal[1] / M - mu * mu;
  float rstd = (float)(1.0 / sqrt(var + (double)LNEPS));
  float muf = (float)mu;
  float v = Cg[idx];
  Ag[idx] = Bg[idx] + (v - muf) * rstd * gma[ch] + bta[ch];
}

__global__ void combine_ffn_kernel(const float* __restrict__ Cg, const double* __restrict__ scal,
                                   const float* __restrict__ gma, const float* __restrict__ bta,
                                   float* __restrict__ Ag) {
  size_t idx = (size_t)blockIdx.x * 256 + threadIdx.x;
  int ch = (int)(idx & 255);
  double M = (double)NNODE * (double)HDIM;
  double mu = scal[0] / M;
  double var = scal[1] / M - mu * mu;
  float rstd = (float)(1.0 / sqrt(var + (double)LNEPS));
  float muf = (float)mu;
  Ag[idx] = (Cg[idx] - muf) * rstd * gma[ch] + bta[ch];
}

// ---------------- fused FFN: gelu(h@W1+b1)@W2+b2 + res; 512-wide tile stays in LDS ----------------
// w1t: bf16 [512 cols][256 k];  w2t: bf16 [256 cols][512 k]
__global__ __launch_bounds__(256) void ffn_kernel(const float* __restrict__ h,
                                                  const unsigned short* __restrict__ w1t,
                                                  const float* __restrict__ b1,
                                                  const unsigned short* __restrict__ w2t,
                                                  const float* __restrict__ b2,
                                                  float* __restrict__ outC, double* __restrict__ scal) {
  __shared__ unsigned short Htl[64 * 264];   // 33792 B
  __shared__ unsigned short Ttl[64 * 520];   // 66560 B
  __shared__ float redA[256], redB[256];
  int tid = threadIdx.x;
  int wave = tid >> 5, lane = tid & 31;
  size_t rowbase = (size_t)blockIdx.x * 64;

  __builtin_prefetch(w1t + (size_t)tid * 512, 0, 1);
  __builtin_prefetch(w2t + (size_t)tid * 512, 0, 1);
  stage_tile_bf16(h + rowbase * HDIM, Htl, 264, tid);
  __syncthreads();

  // ff1 (256 -> 512), wave covers 64 output cols; GELU(exact) -> bf16 LDS
  {
    f32x8 acc[4][4];
#pragma unroll
    for (int rt = 0; rt < 4; ++rt)
#pragma unroll
      for (int j = 0; j < 4; ++j) acc[rt][j] = zero8();
    for (int kc = 0; kc < 8; ++kc) {
      int kb = kc * 32;
      bf16x16 bfr[4];
#pragma unroll
      for (int j = 0; j < 4; ++j) bfr[j] = fragB_bf(w1t, HDIM, kb, wave * 64 + j * 16, lane);
#pragma unroll
      for (int rt = 0; rt < 4; ++rt) {
        bf16x16 a = fragA_lds(Htl, 264, rt * 16, kb, lane);
#pragma unroll
        for (int j = 0; j < 4; ++j) acc[rt][j] = wmma_bf16(a, bfr[j], acc[rt][j]);
      }
    }
#pragma unroll
    for (int rt = 0; rt < 4; ++rt)
#pragma unroll
      for (int j = 0; j < 4; ++j) {
        int col = wave * 64 + j * 16 + (lane & 15);
        float bv = b1[col];
#pragma unroll
        for (int r = 0; r < 8; ++r) {
          int row = rt * 16 + ((lane >> 4) << 3) + r;
          float v = acc[rt][j][r] + bv;
          v = 0.5f * v * (1.f + erff(v * 0.70710678118654752f));
          Ttl[row * 520 + col] = f2bf_u(v);
        }
      }
  }
  __syncthreads();

  // ff2 (512 -> 256) + bias + residual; accumulate LN stats
  float s1 = 0.f, s2 = 0.f;
  {
    int col0 = wave * 32;
    f32x8 acc[4][2];
#pragma unroll
    for (int rt = 0; rt < 4; ++rt) { acc[rt][0] = zero8(); acc[rt][1] = zero8(); }
    for (int kc = 0; kc < 16; ++kc) {
      int kb = kc * 32;
      bf16x16 b0 = fragB_bf(w2t, FFD, kb, col0, lane);
      bf16x16 b1f = fragB_bf(w2t, FFD, kb, col0 + 16, lane);
#pragma unroll
      for (int rt = 0; rt < 4; ++rt) {
        bf16x16 a = fragA_lds(Ttl, 520, rt * 16, kb, lane);
        acc[rt][0] = wmma_bf16(a, b0, acc[rt][0]);
        acc[rt][1] = wmma_bf16(a, b1f, acc[rt][1]);
      }
    }
#pragma unroll
    for (int rt = 0; rt < 4; ++rt)
#pragma unroll
      for (int j = 0; j < 2; ++j) {
        int col = col0 + j * 16 + (lane & 15);
        float bv = b2[col];
#pragma unroll
        for (int r = 0; r < 8; ++r) {
          int row = rt * 16 + ((lane >> 4) << 3) + r;
          float v = acc[rt][j][r] + bv + h[(rowbase + row) * HDIM + col];
          outC[(rowbase + row) * HDIM + col] = v;
          s1 += v; s2 += v * v;
        }
      }
  }
  redA[tid] = s1; redB[tid] = s2;
  __syncthreads();
  for (int off = 128; off > 0; off >>= 1) {
    if (tid < off) { redA[tid] += redA[tid + off]; redB[tid] += redB[tid + off]; }
    __syncthreads();
  }
  if (tid == 0) { atomicAdd(&scal[0], (double)redA[0]); atomicAdd(&scal[1], (double)redB[0]); }
}

// ---------------- mean pool ----------------
__global__ void pool_kernel(const float* __restrict__ A, float* __restrict__ out) {
  int g = blockIdx.x;
  int c = threadIdx.x;
  float s = 0.f;
  for (int n = 0; n < SG; ++n) s += A[((size_t)g * SG + n) * HDIM + c];
  out[(size_t)g * HDIM + c] = s * (1.f / (float)SG);
}

// ---------------- host orchestration ----------------
extern "C" void kernel_launch(void* const* d_in, const int* in_sizes, int n_in,
                              void* d_out, int out_size, void* d_ws, size_t ws_size,
                              hipStream_t stream) {
  (void)in_sizes; (void)n_in; (void)out_size;
  const float* x        = (const float*)d_in[0];
  const float* rwse     = (const float*)d_in[1];
  const int*   eidx     = (const int*)d_in[2];
  // d_in[3] = graph_size (==64, baked into SG)
  const float* emb_x_w  = (const float*)d_in[4];
  const float* emb_x_b  = (const float*)d_in[5];
  const float* bn_g     = (const float*)d_in[6];
  const float* bn_b     = (const float*)d_in[7];
  const float* emb_se_w = (const float*)d_in[8];
  const float* emb_se_b = (const float*)d_in[9];
  const float* gcn_w    = (const float*)d_in[10];
  const float* gcn_b    = (const float*)d_in[11];
  const float* cn_g     = (const float*)d_in[12];
  const float* cn_b     = (const float*)d_in[13];
  const float* q_w      = (const float*)d_in[14];
  const float* q_b      = (const float*)d_in[15];
  const float* k_w      = (const float*)d_in[16];
  const float* k_b      = (const float*)d_in[17];
  const float* o_w      = (const float*)d_in[18];
  const float* o_b      = (const float*)d_in[19];
  const float* an_g     = (const float*)d_in[20];
  const float* an_b     = (const float*)d_in[21];
  const float* ff1_w    = (const float*)d_in[22];
  const float* ff1_b    = (const float*)d_in[23];
  const float* ff2_w    = (const float*)d_in[24];
  const float* ff2_b    = (const float*)d_in[25];
  const float* fn_g     = (const float*)d_in[26];
  const float* fn_b     = (const float*)d_in[27];

  const size_t NHsz = (size_t)NNODE * HDIM;
  const size_t HH   = (size_t)HDIM * HDIM;
  const size_t HF   = (size_t)HDIM * FFD;

  float* Ab   = (float*)d_ws;          // h
  float* Bb   = Ab + NHsz;             // hg (post graph-LN)
  float* Cb   = Bb + NHsz;             // scratch: hw / ha / hf
  float* dinv = Cb + NHsz;             // [N]
  double* scal = (double*)(dinv + NNODE);
  unsigned short* wt_gcn = (unsigned short*)(scal + 2);   // [L][256][256] bf16 (transposed)
  unsigned short* wt_q   = wt_gcn + (size_t)LAYERS * HH;
  unsigned short* wt_k   = wt_q   + (size_t)LAYERS * HH;
  unsigned short* wt_o   = wt_k   + (size_t)LAYERS * HH;
  unsigned short* wt_f1  = wt_o   + (size_t)LAYERS * HH;  // [L][512][256]
  unsigned short* wt_f2  = wt_f1  + (size_t)LAYERS * HF;  // [L][256][512]
  (void)ws_size;                        // requires ~409 MB of d_ws

  const int* e_src = eidx;
  const int* e_dst = eidx + NEDGE;

  // one-shot weight convert+transpose to bf16 (L2-resident thereafter)
  wtconv_kernel<<<(unsigned)(LAYERS * HH / 256), 256, 0, stream>>>(gcn_w, wt_gcn, HDIM, HDIM);
  wtconv_kernel<<<(unsigned)(LAYERS * HH / 256), 256, 0, stream>>>(q_w, wt_q, HDIM, HDIM);
  wtconv_kernel<<<(unsigned)(LAYERS * HH / 256), 256, 0, stream>>>(k_w, wt_k, HDIM, HDIM);
  wtconv_kernel<<<(unsigned)(LAYERS * HH / 256), 256, 0, stream>>>(o_w, wt_o, HDIM, HDIM);
  wtconv_kernel<<<(unsigned)(LAYERS * HF / 256), 256, 0, stream>>>(ff1_w, wt_f1, HDIM, FFD);
  wtconv_kernel<<<(unsigned)(LAYERS * HF / 256), 256, 0, stream>>>(ff2_w, wt_f2, FFD, HDIM);

  deg_init_kernel<<<NNODE / 256, 256, 0, stream>>>(dinv);
  deg_acc_kernel<<<NEDGE / 256, 256, 0, stream>>>(e_dst, dinv);
  deg_fin_kernel<<<NNODE / 256, 256, 0, stream>>>(dinv);

  encoder_kernel<<<(unsigned)(NHsz / 256), 256, 0, stream>>>(
      x, rwse, emb_x_w, emb_x_b, bn_g, bn_b, emb_se_w, emb_se_b, Ab);

  for (int l = 0; l < LAYERS; ++l) {
    // GCN branch: hw = h @ W  -> aggregate + bias + res + graph-LN -> B
    gemm_h_kernel<<<NNODE / 64, 256, 0, stream>>>(Ab, wt_gcn + (size_t)l * HH, Cb);
    gcn_agg_kernel<<<GG, 256, 0, stream>>>(Cb, Ab, e_src, e_dst, dinv,
                                           gcn_b + l * HDIM, cn_g + l * HDIM, cn_b + l * HDIM, Bb);

    // Attention branch: ha = MHA(h) + res -> C, with full-LN stats
    zero2_kernel<<<1, 1, 0, stream>>>(scal);
    mha_kernel<<<GG, 256, 0, stream>>>(Ab, wt_q + (size_t)l * HH, q_b + l * HDIM,
                                       wt_k + (size_t)l * HH, k_b + l * HDIM,
                                       wt_o + (size_t)l * HH, o_b + l * HDIM, Cb, scal);
    combine_attn_kernel<<<(unsigned)(NHsz / 256), 256, 0, stream>>>(
        Bb, Cb, scal, an_g + l * HDIM, an_b + l * HDIM, Ab);

    // FFN: hf = gelu(h@W1+b1)@W2+b2 + res -> C, then full-LN -> h
    zero2_kernel<<<1, 1, 0, stream>>>(scal);
    ffn_kernel<<<NNODE / 64, 256, 0, stream>>>(Ab, wt_f1 + (size_t)l * HF, ff1_b + l * FFD,
                                               wt_f2 + (size_t)l * HF, ff2_b + l * HDIM, Cb, scal);
    combine_ffn_kernel<<<(unsigned)(NHsz / 256), 256, 0, stream>>>(
        Cb, scal, fn_g + l * HDIM, fn_b + l * HDIM, Ab);
  }

  pool_kernel<<<GG, 256, 0, stream>>>(Ab, (float*)d_out);
}
